// Hebbian_13065290514793
// MI455X (gfx1250) — matrix operand order repaired and checked by reading, not compiled, not run on hardware
//
#include <hip/hip_runtime.h>

typedef __bf16 bf16_t;
typedef __attribute__((ext_vector_type(2)))  __bf16 v2bf;
typedef __attribute__((ext_vector_type(16))) __bf16 v16bf;
typedef __attribute__((ext_vector_type(8)))  float  v8f;
typedef __attribute__((ext_vector_type(4)))  int    v4i;

#define N_SEQ 32
#define KDIM  128
#define VDIM  128
#define HHEAD 4
#define HK    512   // H*K
#define HV    512   // H*V

__device__ __forceinline__ bf16_t f2bf(float f) {
#if __has_builtin(__builtin_amdgcn_cvt_pk_bf16_f32)
  v2bf p = __builtin_amdgcn_cvt_pk_bf16_f32(f, 0.0f);
  return p[0];
#else
  unsigned u = __builtin_bit_cast(unsigned, f);
  u += 0x7fffu + ((u >> 16) & 1u);                 // round-to-nearest-even
  unsigned short h = (unsigned short)(u >> 16);
  return __builtin_bit_cast(bf16_t, h);
#endif
}

__device__ __forceinline__ bf16_t bfzero() {
  return __builtin_bit_cast(bf16_t, (unsigned short)0);
}

__device__ __forceinline__ v8f wmma_bf16(v16bf a, v16bf b, v8f c) {
  // 8 args: (neg_a, A, neg_b, B, c_mod, C, reuse_a, reuse_b)
  return __builtin_amdgcn_wmma_f32_16x16x32_bf16(false, a, false, b, (short)0, c,
                                                 false, false);
}

// A fragment (16x32, bf16) from row-major src: A(m,k) = s[(m0+m)*ld + k0+k]
// Lane layout (ISA 7.12.2, 16-bit A 16x32): lane L -> M=L&15; VGPR j holds
// K = 2j + (j>=4 ? 8 : 0) + 8*(L>=16), pairs (K, K+1).
__device__ __forceinline__ v16bf load_a(const bf16_t* s, int ld, int m0, int k0, int lane) {
  int m = lane & 15, half = lane >> 4;
  v16bf a;
#pragma unroll
  for (int j = 0; j < 8; ++j) {
    int kb = k0 + 2 * j + ((j >= 4) ? 8 : 0) + 8 * half;
    const bf16_t* p = s + (m0 + m) * ld + kb;
    a[2 * j]     = p[0];
    a[2 * j + 1] = p[1];
  }
  return a;
}

// Transposed A fragment: logical A(m,k) = s[k*ld + m0+m]  (K = 32 source rows)
__device__ __forceinline__ v16bf load_aT(const bf16_t* s, int ld, int m0, int lane) {
  int m = lane & 15, half = lane >> 4;
  v16bf a;
#pragma unroll
  for (int j = 0; j < 8; ++j) {
    int kb = 2 * j + ((j >= 4) ? 8 : 0) + 8 * half;
    a[2 * j]     = s[kb * ld + m0 + m];
    a[2 * j + 1] = s[(kb + 1) * ld + m0 + m];
  }
  return a;
}

// B fragment (32x16, bf16) from row-major src: B(k,n) = s[(k0+k)*ld + n0+n]
// Lane layout: N = L&15; VGPR j holds K = 2j + (j>=4 ? 8 : 0) + 8*(L>=16), pairs (K, K+1).
__device__ __forceinline__ v16bf load_b(const bf16_t* s, int ld, int k0, int n0, int lane) {
  int n  = lane & 15;
  int half = lane >> 4;
  v16bf b;
#pragma unroll
  for (int j = 0; j < 8; ++j) {
    int kb = k0 + 2 * j + ((j >= 4) ? 8 : 0) + 8 * half;
    const bf16_t* p = s + kb * ld + n0 + n;
    b[2 * j]     = p[0];
    b[2 * j + 1] = p[ld];
  }
  return b;
}

// ---------------------------------------------------------------------------
// K0a: f32 -> bf16 weight conversion
__global__ void cvt_kernel(const float* __restrict__ s, bf16_t* __restrict__ d, int n) {
  int i = blockIdx.x * blockDim.x + threadIdx.x;
  if (i < n) d[i] = f2bf(s[i]);
}

// ---------------------------------------------------------------------------
// K0b: q = query(1024x128) @ Wq(128x512) + bq  -> bf16
__global__ __launch_bounds__(256) void qgemm(const float* __restrict__ query,
                                             const bf16_t* __restrict__ Wq_bf,
                                             const float* __restrict__ bq,
                                             bf16_t* __restrict__ q_bf) {
  int wave = threadIdx.x >> 5, lane = threadIdx.x & 31;
  int gt = blockIdx.x * 8 + wave;          // 2048 tiles: 64 M-tiles x 32 N-tiles
  int m0 = (gt >> 5) * 16, n0 = (gt & 31) * 16;
  int nlane = lane & 15, half = lane >> 4;
  v8f acc = {};
#pragma unroll
  for (int k0 = 0; k0 < KDIM; k0 += 32) {
    v16bf a;
#pragma unroll
    for (int j = 0; j < 8; ++j) {
      int kb = k0 + 2 * j + ((j >= 4) ? 8 : 0) + 8 * half;
      const float* p = query + (m0 + nlane) * KDIM + kb;
      a[2 * j] = f2bf(p[0]);
      a[2 * j + 1] = f2bf(p[1]);
    }
    acc = wmma_bf16(a, load_b(Wq_bf, HK, k0, n0, lane), acc);
  }
  float bias = bq[n0 + nlane];
#pragma unroll
  for (int r = 0; r < 8; ++r) {
    int m = r + 8 * half;
    q_bf[(size_t)(m0 + m) * HK + n0 + nlane] = f2bf(acc[r] + bias);
  }
}

// ---------------------------------------------------------------------------
// K1: per-batch Hebbian write + read numerator. One 8-wave block per batch.
__global__ __launch_bounds__(256) void hebbian_main(
    const float* __restrict__ value, const float* __restrict__ key,
    const float* __restrict__ modulation, const float* __restrict__ w_assoc,
    const int* __restrict__ done_mask,
    const float* __restrict__ bk, const float* __restrict__ bv,
    const float* __restrict__ Amat, const float* __restrict__ Bmat,
    const bf16_t* __restrict__ Wk_bf, const bf16_t* __restrict__ Wv_bf,
    const bf16_t* __restrict__ q_bf,
    float* __restrict__ w_out, bf16_t* __restrict__ read_bf) {
  __shared__ bf16_t sKey[N_SEQ * KDIM];   //  8 KB  raw key (bf16)
  __shared__ bf16_t sVal[N_SEQ * VDIM];   //  8 KB  raw value (bf16)
  __shared__ bf16_t sK[N_SEQ * KDIM];     //  8 KB  encoded k
  __shared__ bf16_t sV[N_SEQ * VDIM];     //  8 KB  encoded v * modulation
  __shared__ bf16_t sReg[KDIM * KDIM];    // 32 KB  k^T k
  __shared__ bf16_t sW[KDIM * VDIM];      // 32 KB  w_assoc (bf16)
  __shared__ bf16_t sWn[KDIM * VDIM];     // 32 KB  w_new (bf16)
  __shared__ float  sWf[KDIM * VDIM];     // 64 KB  w_assoc (f32)
  __shared__ bf16_t sQ[HK];               //  1 KB  q heads (4x128)

  const int b = blockIdx.x;
  const int tid = threadIdx.x;
  const int wave = tid >> 5;
  const int lane = tid & 31;
  const int nlane = lane & 15, half = lane >> 4;

  // ---- stage (async DMA for the 64KB w tile; coalesced loads for the rest) ----
  const float* gw = w_assoc + (size_t)b * KDIM * VDIM;
#if __has_builtin(__builtin_amdgcn_global_load_async_to_lds_b128)
  for (int i = tid * 4; i < KDIM * VDIM; i += 256 * 4) {
    __builtin_amdgcn_global_load_async_to_lds_b128(
        (__attribute__((address_space(1))) v4i*)(gw + i),
        (__attribute__((address_space(3))) v4i*)(&sWf[i]), 0, 0);
  }
#else
  for (int i = tid; i < KDIM * VDIM; i += 256) sWf[i] = gw[i];
#endif
  for (int i = tid; i < N_SEQ * KDIM; i += 256) {
    sKey[i] = f2bf(key[(size_t)b * N_SEQ * KDIM + i]);
    sVal[i] = f2bf(value[(size_t)b * N_SEQ * VDIM + i]);
  }
  for (int i = tid; i < HK; i += 256) sQ[i] = q_bf[(size_t)b * HK + i];
#if __has_builtin(__builtin_amdgcn_s_wait_asynccnt)
  __builtin_amdgcn_s_wait_asynccnt(0);
#else
  asm volatile("s_wait_asynccnt 0" ::: "memory");
#endif
  __syncthreads();
  // bf16 shadow of w (LDS -> LDS, after all async writes are visible)
  for (int i = tid; i < KDIM * VDIM; i += 256) sW[i] = f2bf(sWf[i]);
  __syncthreads();

  // ---- Phase A: k = key@Wk + bk ; v = (value@Wv + bv)*mod ----
  for (int t = wave; t < 16; t += 8) {        // 2x8 tiles of 16x16
    int m0 = (t >> 3) * 16;
    int n0 = (t & 7) * 16;
    v8f ak = {}, av = {};
#pragma unroll
    for (int k0 = 0; k0 < KDIM; k0 += 32) {
      ak = wmma_bf16(load_a(sKey, KDIM, m0, k0, lane),
                     load_b(Wk_bf, KDIM, k0, n0, lane), ak);
      av = wmma_bf16(load_a(sVal, VDIM, m0, k0, lane),
                     load_b(Wv_bf, VDIM, k0, n0, lane), av);
    }
    float biask = bk[n0 + nlane];
    float biasv = bv[n0 + nlane];
#pragma unroll
    for (int r = 0; r < 8; ++r) {
      int row = m0 + r + 8 * half;
      float md = modulation[b * N_SEQ + row];
      sK[row * KDIM + n0 + nlane] = f2bf(ak[r] + biask);
      sV[row * VDIM + n0 + nlane] = f2bf((av[r] + biasv) * md);
    }
  }
  __syncthreads();

  // ---- Phase B: reg = k^T @ k (wave w owns row-tile kk0 = 16w) ----
  {
    int kk0 = wave * 16;
    v16bf aT = load_aT(sK, KDIM, kk0, lane);
#pragma unroll
    for (int j = 0; j < 8; ++j) {
      int l0 = j * 16;
      v8f acc = {};
      acc = wmma_bf16(aT, load_b(sK, KDIM, 0, l0, lane), acc);
#pragma unroll
      for (int r = 0; r < 8; ++r)
        sReg[(kk0 + r + 8 * half) * KDIM + l0 + nlane] = f2bf(acc[r]);
    }
  }
  __syncthreads();

  // ---- Phase C: corr, reg@w, fused elementwise w_new (stream f32 out) ----
  {
    int kk0 = wave * 16;
    const float maskf = (float)done_mask[b];
    v16bf aT = load_aT(sK, KDIM, kk0, lane);       // k^T fragment, reused
#pragma unroll
    for (int j = 0; j < 8; ++j) {
      int vv0 = j * 16;
      v8f accC = {};
      accC = wmma_bf16(aT, load_b(sV, VDIM, 0, vv0, lane), accC);
      v8f accR = {};
#pragma unroll
      for (int k0 = 0; k0 < KDIM; k0 += 32)
        accR = wmma_bf16(load_a(sReg, KDIM, kk0, k0, lane),
                         load_b(sW, VDIM, k0, vv0, lane), accR);
#pragma unroll
      for (int r = 0; r < 8; ++r) {
        int idx = (kk0 + r + 8 * half) * VDIM + vv0 + nlane;
        float w0 = sWf[idx];
        float nw = w0 + maskf * (Amat[idx] * (1.0f - w0) * accC[r]
                                 - Bmat[idx] * accR[r]);
        w_out[(size_t)b * KDIM * VDIM + idx] = nw;
        sWn[idx] = f2bf(nw);
      }
    }
  }
  __syncthreads();

  // ---- Phase D: read = q(4x128, padded to 16) @ w_new ----
  {
    int n0 = wave * 16;
    v8f acc = {};
#pragma unroll
    for (int k0 = 0; k0 < KDIM; k0 += 32) {
      v16bf a;
#pragma unroll
      for (int j = 0; j < 8; ++j) {
        int kb = k0 + 2 * j + ((j >= 4) ? 8 : 0) + 8 * half;
        if (nlane < HHEAD) {
          a[2 * j] = sQ[nlane * KDIM + kb];
          a[2 * j + 1] = sQ[nlane * KDIM + kb + 1];
        } else {
          a[2 * j] = bfzero();
          a[2 * j + 1] = bfzero();
        }
      }
      acc = wmma_bf16(a, load_b(sWn, VDIM, k0, n0, lane), acc);
    }
#pragma unroll
    for (int r = 0; r < 8; ++r) {
      int m = r + 8 * half;
      if (m < HHEAD)
        read_bf[(size_t)b * HV + m * VDIM + n0 + nlane] = f2bf(acc[r]);
    }
  }
}

// ---------------------------------------------------------------------------
// K2: out = read(1024x512) @ Wagg(512x128) + bagg
__global__ __launch_bounds__(256) void outgemm(const bf16_t* __restrict__ read_bf,
                                               const bf16_t* __restrict__ Wagg_bf,
                                               const float* __restrict__ bagg,
                                               float* __restrict__ out) {
  int wave = threadIdx.x >> 5, lane = threadIdx.x & 31;
  int m0 = blockIdx.x * 16, n0 = wave * 16;
  int nlane = lane & 15, half = lane >> 4;
  v8f acc = {};
#pragma unroll 4
  for (int k0 = 0; k0 < HV; k0 += 32)
    acc = wmma_bf16(load_a(read_bf, HV, m0, k0, lane),
                    load_b(Wagg_bf, VDIM, k0, n0, lane), acc);
  float bias = bagg[n0 + nlane];
#pragma unroll
  for (int r = 0; r < 8; ++r) {
    int m = r + 8 * half;
    out[(size_t)(m0 + m) * VDIM + n0 + nlane] = acc[r] + bias;
  }
}

// ---------------------------------------------------------------------------
extern "C" void kernel_launch(void* const* d_in, const int* in_sizes, int n_in,
                              void* d_out, int out_size, void* d_ws, size_t ws_size,
                              hipStream_t stream) {
  const float* value      = (const float*)d_in[0];
  const float* key        = (const float*)d_in[1];
  const float* modulation = (const float*)d_in[2];
  const float* query      = (const float*)d_in[3];
  const float* w_assoc    = (const float*)d_in[4];
  const int*   done_mask  = (const int*)d_in[5];
  const float* Wk         = (const float*)d_in[6];
  const float* bk         = (const float*)d_in[7];
  const float* Wv         = (const float*)d_in[8];
  const float* bv         = (const float*)d_in[9];
  const float* Amat       = (const float*)d_in[10];
  const float* Bmat       = (const float*)d_in[11];
  const float* Wq         = (const float*)d_in[12];
  const float* bq         = (const float*)d_in[13];
  const float* Wagg       = (const float*)d_in[14];
  const float* bagg       = (const float*)d_in[15];

  float* out   = (float*)d_out;        // [1024*128]
  float* w_out = out + 1024 * 128;     // [1024*128*128] (second tuple element)

  char* ws = (char*)d_ws;
  bf16_t* Wk_bf   = (bf16_t*)(ws);                              //  32 KB
  bf16_t* Wv_bf   = (bf16_t*)(ws + 32768);                      //  32 KB
  bf16_t* Wq_bf   = (bf16_t*)(ws + 65536);                      // 128 KB
  bf16_t* Wagg_bf = (bf16_t*)(ws + 65536 + 131072);             // 128 KB
  bf16_t* q_bf    = (bf16_t*)(ws + 65536 + 262144);             //   1 MB
  bf16_t* read_bf = (bf16_t*)(ws + 65536 + 262144 + 1048576);   //   1 MB

  cvt_kernel<<<64, 256, 0, stream>>>(Wk, Wk_bf, 16384);
  cvt_kernel<<<64, 256, 0, stream>>>(Wv, Wv_bf, 16384);
  cvt_kernel<<<256, 256, 0, stream>>>(Wq, Wq_bf, 65536);
  cvt_kernel<<<256, 256, 0, stream>>>(Wagg, Wagg_bf, 65536);
  qgemm<<<256, 256, 0, stream>>>(query, Wq_bf, bq, q_bf);
  hebbian_main<<<1024, 256, 0, stream>>>(value, key, modulation, w_assoc, done_mask,
                                         bk, bv, Amat, Bmat, Wk_bf, Wv_bf, q_bf,
                                         w_out, read_bf);
  outgemm<<<64, 256, 0, stream>>>(read_bf, Wagg_bf, bagg, out);
}
